// Block_59141699666119
// MI455X (gfx1250) — compile-verified, matrix operand order checked by reading, not run on hardware
//
#include <hip/hip_runtime.h>
#include <hip/hip_bf16.h>
#include <math.h>

// ---------------- CDNA5 WMMA helpers (wave32, 16x16x32 f16 -> f32) ----------

typedef __attribute__((ext_vector_type(16))) _Float16 v16h;
typedef __attribute__((ext_vector_type(8)))  float    v8f;
typedef __attribute__((ext_vector_type(4)))  unsigned int u32x4;
typedef __attribute__((ext_vector_type(8)))  int      i32x8;
typedef __attribute__((ext_vector_type(4)))  int      i32x4;

union FragU { v16h v; int4 q[2]; };
union H8U   { _Float16 h[8]; int4 q; };

// Load a 16x32 (rows x K) f16 fragment from a row-major tile with leading dim
// `ld` (elements). Per ISA dense 16-bit A layout: lane m (0-15) holds row m,
// K = {0..7,16..23}; lane m+16 holds row m, K = {8..15,24..31}.
// B operand of an NT GEMM (K-major B^T rows) uses the identical pattern with
// lane = output column.
__device__ __forceinline__ v16h load_frag(const _Float16* base, int ld) {
  const int lane = threadIdx.x & 31;
  const _Float16* p = base + (lane & 15) * ld + ((lane >> 4) << 3);
  FragU f;
  f.q[0] = *(const int4*)(p);
  f.q[1] = *(const int4*)(p + 16);
  return f.v;
}

__device__ __forceinline__ v8f wmma_f16(v16h a, v16h b, v8f c) {
  return __builtin_amdgcn_wmma_f32_16x16x32_f16(false, a, false, b,
                                                (short)0, c, false, false);
}

// ---------------- Tensor Data Mover: 2-D f16 tile -> padded LDS rows --------
// D# per CDNA5 ISA ch.8: group0 = {count|flags, lds_addr, global_addr, type=2},
// group1 = {data_size=2B, pad_enable, pad_interval(16 dw), pad_amount(4 dw),
//           tensor dims, tile dims, dim0 stride}. Tile rows are `tile0` f16
// (64B) landing at 80B (LDT=40) LDS pitch via the pad fields.
#define LDT 40

__device__ __forceinline__ void tdm_load_2d(unsigned lds_byte_off,
                                            const _Float16* gptr,
                                            unsigned tdim0, unsigned tdim1,
                                            unsigned tile0, unsigned tile1,
                                            unsigned stride0) {
  unsigned long long ga = (unsigned long long)gptr;
  u32x4 g0 = {};
  g0[0] = 1u;                                   // count=1, user mode
  g0[1] = lds_byte_off;                         // LDS byte address
  g0[2] = (unsigned)ga;                         // global_addr[31:0]
  g0[3] = (unsigned)((ga >> 32) & 0x01ffffffu)  // global_addr[56:32]
        | 0x80000000u;                          // type=2 ("image")
  i32x8 g1 = {};
  g1[0] = (int)((1u << 16)        // data_size = 2 bytes
              | (1u << 20)        // pad_enable
              | (3u << 22)        // pad_interval: 16 DWORDs (64B)
              | (3u << 25));      // pad_amount:   4 DWORDs (16B)
  g1[1] = (int)((tdim0 & 0xffffu) << 16);                    // tensor_dim0 lo
  g1[2] = (int)((tdim0 >> 16) | ((tdim1 & 0xffffu) << 16));  // d0 hi | d1 lo
  g1[3] = (int)((tdim1 >> 16) | (tile0 << 16));              // d1 hi | tile0
  g1[4] = (int)(tile1 & 0xffffu);                            // tile1, tile2=0
  g1[5] = (int)stride0;                                      // dim0 stride lo
  g1[6] = 0;
  g1[7] = 0;
  i32x4 gz = {};
#if defined(__clang_major__) && __clang_major__ >= 23
  i32x8 gz8 = {};
  __builtin_amdgcn_tensor_load_to_lds(g0, g1, gz, gz, gz8, 0);
#else
  __builtin_amdgcn_tensor_load_to_lds(g0, g1, gz, gz, 0);
#endif
}

// ---------------- f32 -> f16 weight conversion ------------------------------

__global__ __launch_bounds__(256)
void cvt_kernel(const float* __restrict__ in, _Float16* __restrict__ out) {
  const int i = (blockIdx.x * 256 + threadIdx.x) * 8;
  float4 a = *(const float4*)(in + i);
  float4 b = *(const float4*)(in + i + 4);
  H8U o;
  o.h[0] = (_Float16)a.x; o.h[1] = (_Float16)a.y;
  o.h[2] = (_Float16)a.z; o.h[3] = (_Float16)a.w;
  o.h[4] = (_Float16)b.x; o.h[5] = (_Float16)b.y;
  o.h[6] = (_Float16)b.z; o.h[7] = (_Float16)b.w;
  *(int4*)(out + i) = o.q;
}

// ---------------- LayerNorm: fp32 in -> f16 normalized out ------------------

__global__ __launch_bounds__(256)
void ln_kernel(const float* __restrict__ x, const float* __restrict__ w,
               const float* __restrict__ b, _Float16* __restrict__ out) {
  __shared__ float rs[8], rss[8];
  const int row = blockIdx.x;
  const int tid = threadIdx.x;
  const float* xr = x + row * 2048;
  float4 v0 = *(const float4*)(xr + tid * 8);
  float4 v1 = *(const float4*)(xr + tid * 8 + 4);
  float s  = v0.x + v0.y + v0.z + v0.w + v1.x + v1.y + v1.z + v1.w;
  float ss = v0.x*v0.x + v0.y*v0.y + v0.z*v0.z + v0.w*v0.w
           + v1.x*v1.x + v1.y*v1.y + v1.z*v1.z + v1.w*v1.w;
#pragma unroll
  for (int off = 16; off >= 1; off >>= 1) {
    s  += __shfl_xor(s,  off, 32);
    ss += __shfl_xor(ss, off, 32);
  }
  if ((tid & 31) == 0) { rs[tid >> 5] = s; rss[tid >> 5] = ss; }
  __syncthreads();
  float tot = 0.f, tots = 0.f;
#pragma unroll
  for (int i = 0; i < 8; ++i) { tot += rs[i]; tots += rss[i]; }
  const float mu   = tot * (1.f / 2048.f);
  const float var  = tots * (1.f / 2048.f) - mu * mu;
  const float rstd = rsqrtf(var + 1e-5f);
  const float in[8] = {v0.x, v0.y, v0.z, v0.w, v1.x, v1.y, v1.z, v1.w};
  const float* wv = w + tid * 8;
  const float* bv = b + tid * 8;
  H8U oo;
#pragma unroll
  for (int i = 0; i < 8; ++i)
    oo.h[i] = (_Float16)((in[i] - mu) * rstd * wv[i] + bv[i]);
  *(int4*)(out + row * 2048 + tid * 8) = oo.q;
}

// ---------------- Generic NT GEMM: C = A(f16,[M,K]) * W(f16,[N,K])^T --------
// Block tile 128x64, 8 waves of 32x32, K-step 32. Double-buffered LDS fed by
// the Tensor Data Mover (wave 0 issues, all waves consume after the barrier).

__global__ __launch_bounds__(256)
void gemm_nt(const _Float16* __restrict__ A, const _Float16* __restrict__ W,
             const float* __restrict__ bias, int M, int N, int K, int mode,
             const float* resid, float* outf, _Float16* __restrict__ outh,
             _Float16* __restrict__ qb, _Float16* __restrict__ kb,
             _Float16* __restrict__ vtb) {
  __shared__ _Float16 As[2][128][LDT];
  __shared__ _Float16 Bs[2][64][LDT];
  const int tid = threadIdx.x;
  const int m0 = blockIdx.y * 128;
  const int n0 = blockIdx.x * 64;
  const int wid   = tid >> 5;
  const int mbase = (wid & 3) * 32;
  const int nbase = (wid >> 2) * 32;

  v8f acc00 = {}, acc01 = {}, acc10 = {}, acc11 = {};
  const int nsteps = K >> 5;

  if (wid == 0) {  // prologue: TDM fills bank 0 for k0 = 0
    tdm_load_2d((unsigned)(unsigned long long)&As[0][0][0],
                A + m0 * K, (unsigned)K, (unsigned)M, 32u, 128u, (unsigned)K);
    tdm_load_2d((unsigned)(unsigned long long)&Bs[0][0][0],
                W + n0 * K, (unsigned)K, (unsigned)N, 32u, 64u, (unsigned)K);
  }

  int buf = 0;
  for (int i = 0; i < nsteps; ++i) {
    if (wid == 0) __builtin_amdgcn_s_wait_tensorcnt(0);  // bank `buf` ready
    __syncthreads();  // publish TDM data; everyone done reading bank buf^1
    if (wid == 0 && i + 1 < nsteps) {  // overlap next DMA with this compute
      const int kn = (i + 1) << 5;
      tdm_load_2d((unsigned)(unsigned long long)&As[buf ^ 1][0][0],
                  A + m0 * K + kn, (unsigned)K, (unsigned)M, 32u, 128u,
                  (unsigned)K);
      tdm_load_2d((unsigned)(unsigned long long)&Bs[buf ^ 1][0][0],
                  W + n0 * K + kn, (unsigned)K, (unsigned)N, 32u, 64u,
                  (unsigned)K);
    }
    v16h a0 = load_frag(&As[buf][mbase][0],      LDT);
    v16h a1 = load_frag(&As[buf][mbase + 16][0], LDT);
    v16h b0 = load_frag(&Bs[buf][nbase][0],      LDT);
    v16h b1 = load_frag(&Bs[buf][nbase + 16][0], LDT);
    acc00 = wmma_f16(a0, b0, acc00);
    acc01 = wmma_f16(a0, b1, acc01);
    acc10 = wmma_f16(a1, b0, acc10);
    acc11 = wmma_f16(a1, b1, acc11);
    buf ^= 1;
  }

  const int lane   = tid & 31;
  const int halfid = lane >> 4;
  const int nn     = lane & 15;
  v8f accs[4] = {acc00, acc01, acc10, acc11};
#pragma unroll
  for (int f = 0; f < 4; ++f) {
    const int fi = f >> 1, fj = f & 1;
#pragma unroll
    for (int r = 0; r < 8; ++r) {
      int m = m0 + mbase + fi * 16 + halfid * 8 + r;
      int n = n0 + nbase + fj * 16 + nn;
      float v = accs[f][r] + bias[n];
      if (mode == 0) {
        // QKV scatter: Q,K row-major [BH,T,128]; V transposed [BH,128,T]
        int region = n >> 11, wv_ = n & 2047;
        int h = wv_ >> 7, d = wv_ & 127;
        int bb = m >> 10, t = m & 1023;
        int bh = bb * 16 + h;
        if (region == 0)      qb [(bh * 1024 + t) * 128 + d]  = (_Float16)v;
        else if (region == 1) kb [(bh * 1024 + t) * 128 + d]  = (_Float16)v;
        else                  vtb[(bh * 128 + d) * 1024 + t]  = (_Float16)v;
      } else if (mode == 2) {
        // fc1 + exact GELU -> f16
        float g = 0.5f * v * (1.0f + erff(v * 0.70710678118654752f));
        outh[m * 8192 + n] = (_Float16)g;
      } else {
        // mode 1 (out-proj) / mode 3 (fc2): + residual -> fp32
        outf[m * 2048 + n] = v + resid[m * 2048 + n];
      }
    }
  }
}

// ---------------- Flash attention: 16 heads, D=128, T=1024, causal ----------
// Block = 4 waves, 64 q rows; each wave owns 16 q rows; K-tiles of 32.

__global__ __launch_bounds__(128)
void attn_kernel(const _Float16* __restrict__ Q, const _Float16* __restrict__ Kd,
                 const _Float16* __restrict__ Vt, _Float16* __restrict__ out) {
  __shared__ _Float16 Plds[4][16][LDT];
  const int tid    = threadIdx.x;
  const int wid    = tid >> 5;
  const int lane   = tid & 31;
  const int halfid = lane >> 4;
  const int nn     = lane & 15;
  const int bh  = blockIdx.y;
  const int qt  = blockIdx.x;
  const int qw0 = qt * 64 + wid * 16;

  const _Float16* Qp = Q + (bh * 1024 + qw0) * 128;
  v16h qa[4];
#pragma unroll
  for (int kk = 0; kk < 4; ++kk) qa[kk] = load_frag(Qp + kk * 32, 128);

  v8f zero = {};
  v8f o[8];
#pragma unroll
  for (int j = 0; j < 8; ++j) o[j] = zero;
  float mr[8], lr[8];
#pragma unroll
  for (int r = 0; r < 8; ++r) { mr[r] = -__builtin_inff(); lr[r] = 0.f; }

  const float scale = 0.08838834764831845f;  // 1/sqrt(128)
  const float L2E   = 1.4426950408889634f;
  const int nkt = qt * 2 + 2;  // uniform across waves -> __syncthreads is safe

  for (int it = 0; it < nkt; ++it) {
    const int k0 = it * 32;
    const _Float16* Kp = Kd + (bh * 1024 + k0) * 128;
    // prefetch next K tile while this one computes (global_prefetch_b8)
    if (it + 1 < nkt) __builtin_prefetch(Kp + 32 * 128 + lane * 64, 0, 1);
    v8f s0 = {}, s1 = {};
#pragma unroll
    for (int kk = 0; kk < 4; ++kk) {
      v16h b0 = load_frag(Kp + kk * 32,            128);
      v16h b1 = load_frag(Kp + 16 * 128 + kk * 32, 128);
      s0 = wmma_f16(qa[kk], b0, s0);
      s1 = wmma_f16(qa[kk], b1, s1);
    }
    float alpha[8];
#pragma unroll
    for (int r = 0; r < 8; ++r) {
      int qrow = qw0 + halfid * 8 + r;
      int kc   = k0 + nn;
      float v0 = (kc      <= qrow) ? s0[r] * scale : -__builtin_inff();
      float v1 = (kc + 16 <= qrow) ? s1[r] * scale : -__builtin_inff();
      float tmax = fmaxf(v0, v1);
#pragma unroll
      for (int off = 8; off >= 1; off >>= 1)
        tmax = fmaxf(tmax, __shfl_xor(tmax, off, 32));
      float mnew = fmaxf(mr[r], tmax);
      float a  = exp2f((mr[r] - mnew) * L2E);
      float p0 = exp2f((v0 - mnew) * L2E);
      float p1 = exp2f((v1 - mnew) * L2E);
      float ps = p0 + p1;
#pragma unroll
      for (int off = 8; off >= 1; off >>= 1) ps += __shfl_xor(ps, off, 32);
      lr[r] = lr[r] * a + ps;
      mr[r] = mnew;
      alpha[r] = a;
      Plds[wid][halfid * 8 + r][nn]      = (_Float16)p0;
      Plds[wid][halfid * 8 + r][nn + 16] = (_Float16)p1;
    }
#pragma unroll
    for (int j = 0; j < 8; ++j)
#pragma unroll
      for (int r = 0; r < 8; ++r) o[j][r] *= alpha[r];
    __syncthreads();
    v16h pa = load_frag(&Plds[wid][0][0], LDT);  // C-layout -> A-layout bounce
    const _Float16* Vp = Vt + (bh * 128) * 1024 + k0;
#pragma unroll
    for (int j = 0; j < 8; ++j) {
      v16h bj = load_frag(Vp + (j * 16) * 1024, 1024);
      o[j] = wmma_f16(pa, bj, o[j]);
    }
    __syncthreads();
  }

  const int bb = bh >> 4, hh = bh & 15;
#pragma unroll
  for (int j = 0; j < 8; ++j)
#pragma unroll
    for (int r = 0; r < 8; ++r) {
      int t = qw0 + halfid * 8 + r;
      int d = j * 16 + nn;
      out[(bb * 1024 + t) * 2048 + hh * 128 + d] = (_Float16)(o[j][r] / lr[r]);
    }
}

// ---------------- Host-side orchestration -----------------------------------

extern "C" void kernel_launch(void* const* d_in, const int* in_sizes, int n_in,
                              void* d_out, int out_size, void* d_ws, size_t ws_size,
                              hipStream_t stream) {
  (void)in_sizes; (void)n_in; (void)out_size; (void)ws_size;
  const float* x     = (const float*)d_in[0];
  const float* ln1_w = (const float*)d_in[1];
  const float* ln1_b = (const float*)d_in[2];
  const float* qkv_w = (const float*)d_in[3];
  const float* qkv_b = (const float*)d_in[4];
  const float* out_w = (const float*)d_in[5];
  const float* out_b = (const float*)d_in[6];
  const float* ln2_w = (const float*)d_in[7];
  const float* ln2_b = (const float*)d_in[8];
  const float* fc1_w = (const float*)d_in[9];
  const float* fc1_b = (const float*)d_in[10];
  const float* fc2_w = (const float*)d_in[11];
  const float* fc2_b = (const float*)d_in[12];
  float* outp = (float*)d_out;  // holds x1 = x + attn after out-proj

  char* ws = (char*)d_ws;
  size_t off = 0;
  _Float16* wqkv = (_Float16*)(ws + off); off += 6144ull * 2048 * 2;
  _Float16* wout = (_Float16*)(ws + off); off += 2048ull * 2048 * 2;
  _Float16* wfc1 = (_Float16*)(ws + off); off += 8192ull * 2048 * 2;
  _Float16* wfc2 = (_Float16*)(ws + off); off += 2048ull * 8192 * 2;
  _Float16* ln1x = (_Float16*)(ws + off); off += 8192ull * 2048 * 2;
  _Float16* qb   = (_Float16*)(ws + off); off += 128ull * 1024 * 128 * 2;
  _Float16* kb   = (_Float16*)(ws + off); off += 128ull * 1024 * 128 * 2;
  _Float16* vtb  = (_Float16*)(ws + off); off += 128ull * 1024 * 128 * 2;
  _Float16* attn = (_Float16*)(ws + off); off += 8192ull * 2048 * 2;
  _Float16* ln2x = (_Float16*)(ws + off); off += 8192ull * 2048 * 2;
  _Float16* hbuf = (_Float16*)(ws + off); off += 8192ull * 8192 * 2;

  // one-time (per launch) weight down-conversion; f16 weights then stay in L2
  cvt_kernel<<<6144, 256, 0, stream>>>(qkv_w, wqkv);
  cvt_kernel<<<2048, 256, 0, stream>>>(out_w, wout);
  cvt_kernel<<<8192, 256, 0, stream>>>(fc1_w, wfc1);
  cvt_kernel<<<8192, 256, 0, stream>>>(fc2_w, wfc2);

  // x -> LN1 -> f16
  ln_kernel<<<8192, 256, 0, stream>>>(x, ln1_w, ln1_b, ln1x);
  // QKV projection, scatter into Q, K (row-major) and V^T
  gemm_nt<<<dim3(96, 64), 256, 0, stream>>>(ln1x, wqkv, qkv_b, 8192, 6144, 2048, 0,
                                            nullptr, nullptr, nullptr, qb, kb, vtb);
  // causal flash attention
  attn_kernel<<<dim3(16, 128), 128, 0, stream>>>(qb, kb, vtb, attn);
  // out projection + residual -> x1 (fp32, in d_out)
  gemm_nt<<<dim3(32, 64), 256, 0, stream>>>(attn, wout, out_b, 8192, 2048, 2048, 1,
                                            x, outp, nullptr, nullptr, nullptr, nullptr);
  // LN2 -> f16
  ln_kernel<<<8192, 256, 0, stream>>>(outp, ln2_w, ln2_b, ln2x);
  // fc1 + GELU -> f16
  gemm_nt<<<dim3(128, 64), 256, 0, stream>>>(ln2x, wfc1, fc1_b, 8192, 8192, 2048, 2,
                                             nullptr, nullptr, hbuf, nullptr, nullptr, nullptr);
  // fc2 + residual, in-place on d_out
  gemm_nt<<<dim3(32, 64), 256, 0, stream>>>(hbuf, wfc2, fc2_b, 8192, 2048, 8192, 3,
                                            outp, outp, nullptr, nullptr, nullptr, nullptr);
}
